// RNN_16484084482357
// MI455X (gfx1250) — matrix operand order repaired and checked by reading, not compiled, not run on hardware
//
#include <hip/hip_runtime.h>
#include <hip/hip_bf16.h>

typedef __attribute__((ext_vector_type(16))) __bf16 bf16x16;
typedef __attribute__((ext_vector_type(8)))  __bf16 bf16x8;
typedef __attribute__((ext_vector_type(8)))  float  f32x8;

#define B_   1024
#define T_   512
#define F_   40
#define H_   50
#define G_   200      // 4*H
#define KPAD 96       // 40 (x) + 50 (h) padded to 3 k-tiles of 32
#define NPAD 256      // 200 gates padded to 16 n-tiles of 16
#define ROWS 16       // batch rows per workgroup (WMMA M)
#define NTILES 13     // ceil(200/16)
#define NTHREADS 416  // 13 waves == 13 gate tiles, no idle waves

// Fast activations: v_exp_f32 + v_rcp_f32 (2 TRANS ops each).
__device__ __forceinline__ float sigmoid_fast(float x) {
  float e = __builtin_amdgcn_exp2f(-1.44269504f * x);
  return __builtin_amdgcn_rcpf(1.f + e);
}
__device__ __forceinline__ float tanh_fast(float x) {
  float e = __builtin_amdgcn_exp2f(-2.88539008f * x);   // exp(-2x)
  return 2.f * __builtin_amdgcn_rcpf(1.f + e) - 1.f;    // 2*sigmoid(2x)-1
}

// ---------------------------------------------------------------------------
// Setup: pack [W_ih | W_hh] into zero-padded bf16 Wc[NPAD][KPAD], gate-major.
// ---------------------------------------------------------------------------
__global__ void pack_weights(const float* __restrict__ W_ih,
                             const float* __restrict__ W_hh,
                             __bf16* __restrict__ wc) {
  int idx = blockIdx.x * blockDim.x + threadIdx.x;
  if (idx >= NPAD * KPAD) return;
  int n = idx / KPAD, k = idx % KPAD;
  float v = 0.f;
  if (n < G_) {
    if (k < F_)            v = W_ih[n * F_ + k];
    else if (k < F_ + H_)  v = W_hh[n * H_ + (k - F_)];
  }
  wc[idx] = (__bf16)v;
}

// ---------------------------------------------------------------------------
// Fused LSTM + head.  One block = 16 batch rows for all 512 timesteps.
// Per step/wave: 6x ds_load_b128 -> 3x v_wmma_f32_16x16x32_bf16 (f32 acc).
// x_{t+1} register double-buffered (x is L2-resident: 84 MB vs 192 MB L2).
// Cell state lives in registers; each thread owns two fixed (row, j) slots.
// ---------------------------------------------------------------------------
__global__ __launch_bounds__(NTHREADS)
void lstm_fused(const float* __restrict__ x,
                const float* __restrict__ b_ih,
                const float* __restrict__ b_hh,
                const __bf16* __restrict__ wc,
                const float* __restrict__ W1,
                const float* __restrict__ b1,
                const float* __restrict__ W2,
                const float* __restrict__ b2,
                float* __restrict__ out) {
  __shared__ __align__(16) __bf16 Abuf[ROWS][KPAD];   // [x_t | h | pad], bf16
  __shared__ float gates[ROWS][NPAD];                  // WMMA gate exchange
  __shared__ float hbuf[ROWS][H_ + 2];                 // hidden state, f32

  const int tid   = threadIdx.x;
  const int wave  = tid >> 5;
  const int lane  = tid & 31;
  const int b0    = blockIdx.x * ROWS;
  const int m     = lane & 15;        // A-row / D-column index
  const int khalf = lane >> 4;        // lane half selects K sub-groups
  const int ks    = khalf * 8;
  const int n0    = wave * 16;        // this wave's gate-column tile

  // ---- init LDS: zero Abuf (h0 = 0 + persistent K padding), h0 ----
  for (int e = tid; e < ROWS * KPAD; e += NTHREADS)
    (&Abuf[0][0])[e] = (__bf16)0.f;
  for (int e = tid; e < ROWS * H_; e += NTHREADS) {
    int r = e / H_, j = e % H_;
    hbuf[r][j] = 0.f;
  }

  // ---- fixed elementwise slots: indices computed ONCE, c in registers ----
  const int r0 = tid / H_, j0 = tid % H_;              // slot 0: always valid
  const int e1 = tid + NTHREADS;
  const bool act1 = (e1 < ROWS * H_);                  // slot 1: tid < 384
  const int r1 = act1 ? e1 / H_ : 0, j1 = act1 ? e1 % H_ : 0;
  float c0 = (j0 == 0) ? 1.f : 0.f;                    // reference: c0[:,0]=1
  float c1 = (j1 == 0) ? 1.f : 0.f;

  // ---- hoist weight B-fragments into registers + per-column bias ----
  bf16x16 Bfrag[3] = {};
  float bias = 0.f;
  {
    const int ncol = n0 + m;
    if (ncol < G_) bias = b_ih[ncol] + b_hh[ncol];
    const unsigned int* wcu = (const unsigned int*)wc;  // 2 bf16 per dword
    #pragma unroll
    for (int kt = 0; kt < 3; ++kt) {
      #pragma unroll
      for (int v = 0; v < 8; ++v) {
        int k = kt * 32 + khalf * 8 + (v >= 4 ? 8 : 0) + 2 * v;
        union { unsigned int u; __bf16 h[2]; } cvt;
        cvt.u = wcu[(ncol * KPAD + k) >> 1];
        Bfrag[kt][2 * v]     = cvt.h[0];
        Bfrag[kt][2 * v + 1] = cvt.h[1];
      }
    }
  }

  // ---- x staging: one float2 per thread (320 active), double-buffered ----
  const bool xact = (tid < ROWS * (F_ / 2));           // 16*20 = 320
  const int  xr   = tid / (F_ / 2);
  const int  xf   = tid % (F_ / 2);                    // float2 index in row
  const float* xptr = x + ((long)(b0 + xr) * T_) * F_ + 2 * xf;
  const float* xnext = xptr + F_;                      // running prefetch ptr
  float2 xreg = make_float2(0.f, 0.f);
  if (xact) xreg = *(const float2*)xptr;               // prefetch t = 0

  __syncthreads();

  // ---- sequential scan over T; two barriers per step ----
  for (int t = 0; t < T_; ++t) {
    if (xact) {
      union { unsigned int u; __bf16 h[2]; } p;
      p.h[0] = (__bf16)xreg.x;
      p.h[1] = (__bf16)xreg.y;
      *(unsigned int*)&Abuf[xr][2 * xf] = p.u;         // one ds_store_b32
      // prefetch next step (clamped ptr, no branch); latency hides behind step
      xreg = *(const float2*)((t + 1 < T_) ? xnext : xptr);
      xnext += F_;
    }
    __syncthreads();   // x_t + h_{t-1} visible to all waves

    // load all three A fragments, then chain WMMAs
    bf16x16 a[3];
    #pragma unroll
    for (int kt = 0; kt < 3; ++kt) {
      bf16x8 lo = *(const bf16x8*)&Abuf[m][kt * 32 + ks];
      bf16x8 hi = *(const bf16x8*)&Abuf[m][kt * 32 + 16 + ks];
      #pragma unroll
      for (int i = 0; i < 8; ++i) { a[kt][i] = lo[i]; a[kt][8 + i] = hi[i]; }
    }
    f32x8 acc;
    #pragma unroll
    for (int r = 0; r < 8; ++r) acc[r] = bias;
    #pragma unroll
    for (int kt = 0; kt < 3; ++kt)
      acc = __builtin_amdgcn_wmma_f32_16x16x32_bf16(
          false, a[kt], false, Bfrag[kt], (short)0, acc, false, false);

    // D layout: lane L, VGPR r -> (m = r + 8*khalf, n = n0 + L%16)
    #pragma unroll
    for (int r = 0; r < 8; ++r) gates[r + 8 * khalf][n0 + m] = acc[r];

    __syncthreads();   // gates visible

    // elementwise LSTM cell update; fixed slots, c in registers
    {
      float gi = sigmoid_fast(gates[r0][j0]);
      float gf = sigmoid_fast(gates[r0][H_ + j0]);
      float gg = tanh_fast(gates[r0][2 * H_ + j0]);
      float go = sigmoid_fast(gates[r0][3 * H_ + j0]);
      c0 = gf * c0 + gi * gg;
      float h = go * tanh_fast(c0);
      hbuf[r0][j0] = h;
      Abuf[r0][F_ + j0] = (__bf16)h;   // next step's WMMA A operand
    }
    if (act1) {
      float gi = sigmoid_fast(gates[r1][j1]);
      float gf = sigmoid_fast(gates[r1][H_ + j1]);
      float gg = tanh_fast(gates[r1][2 * H_ + j1]);
      float go = sigmoid_fast(gates[r1][3 * H_ + j1]);
      c1 = gf * c1 + gi * gg;
      float h = go * tanh_fast(c1);
      hbuf[r1][j1] = h;
      Abuf[r1][F_ + j1] = (__bf16)h;
    }
    // next step's post-x barrier orders the h writes before the WMMA reads
  }
  __syncthreads();

  // ---- MLP head: relu(h @ W1^T + b1) @ W2^T + b2 + x[:, -1, 0] ----
  if (tid < ROWS) {
    int r = tid;
    float o = b2[0];
    #pragma unroll
    for (int q = 0; q < 10; ++q) {
      float s = b1[q];
      for (int j = 0; j < H_; ++j) s += hbuf[r][j] * W1[q * H_ + j];
      s = fmaxf(s, 0.f);
      o += s * W2[q];
    }
    o += x[((long)(b0 + r) * T_ + (T_ - 1)) * F_];
    out[b0 + r] = o;
  }
}

extern "C" void kernel_launch(void* const* d_in, const int* in_sizes, int n_in,
                              void* d_out, int out_size, void* d_ws, size_t ws_size,
                              hipStream_t stream) {
  const float* x    = (const float*)d_in[0];
  const float* W_ih = (const float*)d_in[1];
  const float* W_hh = (const float*)d_in[2];
  const float* b_ih = (const float*)d_in[3];
  const float* b_hh = (const float*)d_in[4];
  const float* W1   = (const float*)d_in[5];
  const float* b1   = (const float*)d_in[6];
  const float* W2   = (const float*)d_in[7];
  const float* b2   = (const float*)d_in[8];
  float* out  = (float*)d_out;
  __bf16* wc  = (__bf16*)d_ws;      // 256*96*2 = 48 KB of workspace

  pack_weights<<<(NPAD * KPAD + 255) / 256, 256, 0, stream>>>(W_ih, W_hh, wc);
  lstm_fused<<<B_ / ROWS, NTHREADS, 0, stream>>>(x, b_ih, b_hh, wc,
                                                 W1, b1, W2, b2, out);
}
